// PTQLinear_49684181680850
// MI455X (gfx1250) — 8-bit matrix operand order repaired and checked by reading, compiled, not run on hardware
//
#include <hip/hip_runtime.h>
#include <hip/hip_bf16.h>

typedef __attribute__((ext_vector_type(8))) int v8i;

#define M_DIM 8192
#define K_DIM 4096
#define N_DIM 4096
#define CAL_ROWS 2048
#define KT_TILES (K_DIM / 64)
#define INT8MAXF 127.0f

__device__ __forceinline__ unsigned quant_pack4(float4 v, float4 s, float rscale) {
    int q0 = (int)rintf(fminf(fmaxf(v.x * s.x * rscale, -INT8MAXF), INT8MAXF));
    int q1 = (int)rintf(fminf(fmaxf(v.y * s.y * rscale, -INT8MAXF), INT8MAXF));
    int q2 = (int)rintf(fminf(fmaxf(v.z * s.z * rscale, -INT8MAXF), INT8MAXF));
    int q3 = (int)rintf(fminf(fmaxf(v.w * s.w * rscale, -INT8MAXF), INT8MAXF));
    return (unsigned)(q0 & 0xFF) | ((unsigned)(q1 & 0xFF) << 8) |
           ((unsigned)(q2 & 0xFF) << 16) | ((unsigned)(q3 & 0xFF) << 24);
}

__global__ __launch_bounds__(256) void colstats_k(
    const float* __restrict__ weight, const float* __restrict__ cal,
    float* __restrict__ smooth, float* __restrict__ inv_smooth) {
    int k = blockIdx.x * 256 + threadIdx.x;
    float amax = 1e-4f;
    for (int m = 0; m < CAL_ROWS; ++m)
        amax = fmaxf(amax, fabsf(cal[(size_t)m * K_DIM + k]));
    float wmax = 1e-4f;
    for (int n = 0; n < N_DIM; ++n)
        wmax = fmaxf(wmax, fabsf(weight[(size_t)n * K_DIM + k]));
    float s = fminf(fmaxf(sqrtf(amax) / sqrtf(wmax), 0.25f), 4.0f);
    smooth[k] = s;
    inv_smooth[k] = 1.0f / s;
}

__global__ __launch_bounds__(256) void wquant_k(
    const float* __restrict__ weight, const float* __restrict__ smooth,
    float* __restrict__ wscale, unsigned* __restrict__ packB) {
    __shared__ float red[256];
    const int n = blockIdx.x;
    const int tid = threadIdx.x;
    const float* wrow = weight + (size_t)n * K_DIM;

    float m = 0.0f;
    for (int k = tid; k < K_DIM; k += 256)
        m = fmaxf(m, fabsf(wrow[k] * smooth[k]));
    red[tid] = m;
    __syncthreads();
    for (int s = 128; s > 0; s >>= 1) {
        if (tid < s) red[tid] = fmaxf(red[tid], red[tid + s]);
        __syncthreads();
    }
    const float scale = fmaxf(red[0] / INT8MAXF, 1e-8f);
    if (tid == 0) wscale[n] = scale;
    const float rs = 1.0f / scale;

    const int nt = n >> 4, nl = n & 15;
    for (int kd = tid; kd < K_DIM / 4; kd += 256) {
        const int k = kd * 4;
        float4 wv = *(const float4*)(wrow + k);
        float4 sv = *(const float4*)(smooth + k);
        unsigned pk = quant_pack4(wv, sv, rs);
        const int kt = k >> 6, ksub = k & 63;
        const int hi = (ksub >> 4) & 1;
        const int v  = ((ksub >> 5) << 2) | ((ksub >> 2) & 3);
        const int lane = (hi << 4) | nl;
        packB[((size_t)(nt * KT_TILES + kt) * 32 + lane) * 8 + v] = pk;
    }
}

__global__ __launch_bounds__(256) void xmax_k(
    const float* __restrict__ x, const float* __restrict__ inv_smooth,
    float* __restrict__ part) {
    __shared__ float red[256];
    size_t g = (size_t)blockIdx.x * 256 + threadIdx.x;
    const size_t stride = (size_t)gridDim.x * 256;
    float m = 0.0f;
    for (size_t i = g; i < (size_t)M_DIM * K_DIM; i += stride) {
        int k = (int)(i & (K_DIM - 1));
        m = fmaxf(m, fabsf(x[i] * inv_smooth[k]));
    }
    red[threadIdx.x] = m;
    __syncthreads();
    for (int s = 128; s > 0; s >>= 1) {
        if (threadIdx.x < s)
            red[threadIdx.x] = fmaxf(red[threadIdx.x], red[threadIdx.x + s]);
        __syncthreads();
    }
    if (threadIdx.x == 0) part[blockIdx.x] = red[0];
}

__global__ __launch_bounds__(256) void xmaxfin_k(
    const float* __restrict__ part, float* __restrict__ inscale) {
    __shared__ float red[256];
    float m = 0.0f;
    for (int i = threadIdx.x; i < 4096; i += 256) m = fmaxf(m, part[i]);
    red[threadIdx.x] = m;
    __syncthreads();
    for (int s = 128; s > 0; s >>= 1) {
        if (threadIdx.x < s)
            red[threadIdx.x] = fmaxf(red[threadIdx.x], red[threadIdx.x + s]);
        __syncthreads();
    }
    if (threadIdx.x == 0) inscale[0] = fmaxf(red[0] / INT8MAXF, 1e-8f);
}

__global__ __launch_bounds__(256) void xquant_k(
    const float* __restrict__ x, const float* __restrict__ inv_smooth,
    const float* __restrict__ inscale, unsigned* __restrict__ packA) {
    const size_t d = (size_t)blockIdx.x * 256 + threadIdx.x;
    const int m  = (int)(d / (K_DIM / 4));
    const int kd = (int)(d % (K_DIM / 4));
    const int k = kd * 4;
    const float rs = 1.0f / inscale[0];
    float4 xv = *(const float4*)(x + (size_t)m * K_DIM + k);
    float4 iv = *(const float4*)(inv_smooth + k);
    unsigned pk = quant_pack4(xv, iv, rs);
    const int mt = m >> 4, ml = m & 15;
    const int kt = k >> 6, ksub = k & 63;
    const int hi = (ksub >> 4) & 1;
    const int v  = ((ksub >> 5) << 2) | ((ksub >> 2) & 3);
    const int lane = (hi << 4) | ml;
    packA[((size_t)(mt * KT_TILES + kt) * 32 + lane) * 8 + v] = pk;
}

__global__ __launch_bounds__(256) void gemm_k(
    const unsigned* __restrict__ packA, const unsigned* __restrict__ packB,
    const float* __restrict__ wscale, const float* __restrict__ bias,
    const float* __restrict__ inscale, float* __restrict__ out) {
    __shared__ int ldsB[2][4 * 256];

    const int tid  = threadIdx.x;
    const int lane = tid & 31;
    const int wv   = tid >> 5;
    const int mt   = blockIdx.y * 8 + wv;
    const int nt0  = blockIdx.x * 4;

    const v8i* Ap = ((const v8i*)packA) + (size_t)mt * KT_TILES * 32 + lane;

    const int e  = tid * 4;
    const int jb = e >> 8;
    const int rb = e & 255;
    const unsigned lds_base = (unsigned)(uintptr_t)(void*)&ldsB[0][0];

    auto load_panel_async = [&](int kt, int buf) {
        const unsigned* gsrc = packB +
            ((size_t)((nt0 + jb) * KT_TILES + kt)) * 256 + rb;
        const unsigned lds_dst = lds_base + (unsigned)((buf * 1024 + e) * 4);
        asm volatile("global_load_async_to_lds_b128 %0, %1, off"
                     :: "v"(lds_dst), "v"((unsigned long long)(uintptr_t)gsrc)
                     : "memory");
    };

    v8i acc[4] = {};

    load_panel_async(0, 0);
    for (int kt = 0; kt < KT_TILES; ++kt) {
        asm volatile("s_wait_asynccnt 0x0" ::: "memory");
        __syncthreads();
        if (kt + 1 < KT_TILES) load_panel_async(kt + 1, (kt + 1) & 1);
        if (kt + 2 < KT_TILES) __builtin_prefetch(Ap + (size_t)(kt + 2) * 32, 0, 1);

        const v8i a = Ap[(size_t)kt * 32];
        const int cur = kt & 1;
#pragma unroll
        for (int j = 0; j < 4; ++j) {
            const v8i b = *(const v8i*)&ldsB[cur][j * 256 + lane * 8];
            acc[j] = __builtin_amdgcn_wmma_i32_16x16x64_iu8(
                true, a, true, b, acc[j], false, false);
        }
    }

    const float is = inscale[0];
    const int m0 = mt * 16 + (lane >> 4) * 8;
#pragma unroll
    for (int j = 0; j < 4; ++j) {
        const int n = (nt0 + j) * 16 + (lane & 15);
        const float sc = is * wscale[n];
        const float bz = bias[n];
#pragma unroll
        for (int r = 0; r < 8; ++r)
            out[(size_t)(m0 + r) * N_DIM + n] = (float)acc[j][r] * sc + bz;
    }
}

extern "C" void kernel_launch(void* const* d_in, const int* in_sizes, int n_in,
                              void* d_out, int out_size, void* d_ws, size_t ws_size,
                              hipStream_t stream) {
    const float* x      = (const float*)d_in[0];
    const float* weight = (const float*)d_in[1];
    const float* bias   = (const float*)d_in[2];
    const float* cal    = (const float*)d_in[3];
    float* out = (float*)d_out;

    char* ws = (char*)d_ws;
    float*    smooth     = (float*)(ws + 0);
    float*    inv_smooth = (float*)(ws + 16384);
    float*    wscale     = (float*)(ws + 32768);
    float*    part       = (float*)(ws + 49152);
    float*    inscale    = (float*)(ws + 65536);
    unsigned* packA      = (unsigned*)(ws + (1 << 20));
    unsigned* packB      = (unsigned*)(ws + (1 << 20) + (size_t)M_DIM * K_DIM);

    colstats_k<<<K_DIM / 256, 256, 0, stream>>>(weight, cal, smooth, inv_smooth);
    wquant_k<<<N_DIM, 256, 0, stream>>>(weight, smooth, wscale, packB);
    xmax_k<<<4096, 256, 0, stream>>>(x, inv_smooth, part);
    xmaxfin_k<<<1, 256, 0, stream>>>(part, inscale);
    xquant_k<<<(M_DIM * (K_DIM / 4)) / 256, 256, 0, stream>>>(x, inv_smooth, inscale, packA);

    dim3 grid(N_DIM / 64, M_DIM / 128);
    gemm_k<<<grid, 256, 0, stream>>>(packA, packB, wscale, bias, inscale, out);
}
